// EnhancedRQGNN_40948218200435
// MI455X (gfx1250) — matrix operand order repaired and checked by reading, not compiled, not run on hardware
//
#include <hip/hip_runtime.h>

#define NN   100000
#define EE   1600000
#define FF   128
#define HH   256
#define GG   64
#define NCLSD 2

typedef __attribute__((ext_vector_type(16))) __bf16         v16bf;
typedef __attribute__((ext_vector_type(8)))  unsigned int   v8u;
typedef __attribute__((ext_vector_type(8)))  float          v8f;

__device__ __forceinline__ unsigned short f2bf(float f) {
  unsigned int u = __builtin_bit_cast(unsigned int, f);
  u += 0x7FFFu + ((u >> 16) & 1u);   // round-to-nearest-even
  return (unsigned short)(u >> 16);
}
__device__ __forceinline__ unsigned int f2bf2(float lo, float hi) {
  return (unsigned int)f2bf(lo) | ((unsigned int)f2bf(hi) << 16);
}

// ---------------------------------------------------------------------------
// Fused WMMA GEMM: out = act( sum_p Xp@Wp + bias ) [+ residual]
// X row-major [rows, KD], W row-major [KD, OUT].
// Block = 256 threads (8 waves), MT row-tiles of 16 per block, each wave owns
// OUT/128 column tiles; all tile indices compile-time (no movrel).
// LDS holds packed bf16 pairs: sA[row][kpair], sBt[col][kpair].
// Requires rows % (16*MT) == 0, KD % 32 == 0.
// ---------------------------------------------------------------------------
template <int OUT, int MT, int NP>
__global__ __launch_bounds__(256) void gemm_wmma(
    const float* __restrict__ X0, const float* __restrict__ W0,
    const float* __restrict__ X1, const float* __restrict__ W1,
    const float* __restrict__ X2, const float* __restrict__ W2,
    int KD, const float* __restrict__ bias, const float* __restrict__ residual,
    float* __restrict__ out, int act)
{
  constexpr int TPW   = OUT / 128;   // col tiles per wave (1 or 2)
  constexpr int AROWS = MT * 16;
  constexpr int WI    = OUT / 16;    // W staging iters (per 256 threads)
  constexpr int NSH   = (OUT == 256) ? 8 : 7;

  __shared__ unsigned int sA[AROWS * 16];   // [row][kpair] packed 2xbf16
  __shared__ unsigned int sBt[OUT * 16];    // [col][kpair] packed 2xbf16

  const int tid   = threadIdx.x;
  const int lane  = tid & 31;
  const int wave  = tid >> 5;
  const int half  = lane >> 4;
  const int row16 = lane & 15;
  const int rowBase = blockIdx.x * AROWS;

  v8f vzero = {0.f, 0.f, 0.f, 0.f, 0.f, 0.f, 0.f, 0.f};
  v8f acc[MT][TPW];
#pragma unroll
  for (int mt = 0; mt < MT; ++mt)
#pragma unroll
    for (int tt = 0; tt < TPW; ++tt) acc[mt][tt] = vzero;

  const float* Xs[3] = {X0, X1, X2};
  const float* Ws[3] = {W0, W1, W2};

#pragma unroll
  for (int p = 0; p < NP; ++p) {
    const float* __restrict__ X = Xs[p];
    const float* __restrict__ W = Ws[p];
    for (int k0 = 0; k0 < KD; k0 += 32) {
      // stage A: AROWS x 32 -> packed pairs, float2 coalesced loads
#pragma unroll
      for (int i = 0; i < MT; ++i) {
        int widx = tid + i * 256;                 // AROWS*16 == MT*256
        int r = widx >> 4, kp = widx & 15;
        const float* xp = X + (rowBase + r) * KD + k0 + 2 * kp;
        sA[widx] = f2bf2(xp[0], xp[1]);
      }
      // stage W transposed: sBt[n*16+kp] = pack(W[k0+2kp][n], W[k0+2kp+1][n])
#pragma unroll
      for (int i = 0; i < WI; ++i) {
        int widx = tid + i * 256;
        int n = widx & (OUT - 1), kp = widx >> NSH;
        const float* wp = W + (k0 + 2 * kp) * OUT + n;
        sBt[n * 16 + kp] = f2bf2(wp[0], wp[OUT]);
      }
      __syncthreads();

      // A fragments (ISA 16-bit 16x32 layout; pair index = kb/2)
      v8u afr[MT];
#pragma unroll
      for (int mt = 0; mt < MT; ++mt) {
        int rb = (mt * 16 + row16) * 16;
#pragma unroll
        for (int v = 0; v < 4; ++v) afr[mt][v] = sA[rb + half * 4 + v];
#pragma unroll
        for (int v = 0; v < 4; ++v) afr[mt][4 + v] = sA[rb + 8 + half * 4 + v];
      }
#pragma unroll
      for (int tt = 0; tt < TPW; ++tt) {
        int n = (wave + tt * 8) * 16 + row16;
        v8u bfr;
#pragma unroll
        for (int v = 0; v < 8; ++v) bfr[v] = sBt[n * 16 + half * 4 + (v & 3) + ((v >> 2) << 3)];
        v16bf b = __builtin_bit_cast(v16bf, bfr);
#pragma unroll
        for (int mt = 0; mt < MT; ++mt) {
          v16bf a = __builtin_bit_cast(v16bf, afr[mt]);
          acc[mt][tt] = __builtin_amdgcn_wmma_f32_16x16x32_bf16(
              false, a, false, b, (short)0, acc[mt][tt], false, false);
        }
      }
      __syncthreads();
    }
  }

  // epilogue per ISA f32 C/D layout: VGPR r holds (M = r + 8*half, N = row16)
#pragma unroll
  for (int mt = 0; mt < MT; ++mt) {
#pragma unroll
    for (int tt = 0; tt < TPW; ++tt) {
      const int col = (wave + tt * 8) * 16 + row16;
      float bv = bias ? bias[col] : 0.0f;
#pragma unroll
      for (int r = 0; r < 8; ++r) {
        int row = rowBase + mt * 16 + r + 8 * half;
        float v = acc[mt][tt][r] + bv;
        if (act == 1)      v = (v > 0.0f) ? v : 0.01f * v;  // leaky
        else if (act == 2) v = (v > 0.0f) ? v : 0.0f;       // relu
        if (residual) v += residual[row * OUT + col];
        out[row * OUT + col] = v;
      }
    }
  }
}

// ---------------------------------------------------------------------------
// Utility / sparse kernels
// ---------------------------------------------------------------------------
__global__ void zero_f32(float* p, size_t n) {
  size_t i = (size_t)blockIdx.x * blockDim.x + threadIdx.x;
  if (i < n) p[i] = 0.0f;
}
__global__ void fill_f32(float* p, float v, size_t n) {
  size_t i = (size_t)blockIdx.x * blockDim.x + threadIdx.x;
  if (i < n) p[i] = v;
}
__global__ void neg_copy(float* dst, const float* src, size_t n) {
  size_t i = (size_t)blockIdx.x * blockDim.x + threadIdx.x;
  if (i < n) dst[i] = -src[i];
}
__global__ void scatter_deg(const int* __restrict__ idx, float* deg, int n) {
  int i = blockIdx.x * blockDim.x + threadIdx.x;
  if (i < n) atomicAdd(&deg[idx[i]], 1.0f);
}
__global__ void rsqrt_deg(const float* __restrict__ deg, float* dis, int n) {
  int i = blockIdx.x * blockDim.x + threadIdx.x;
  if (i < n) {
    float d = deg[i];
    dis[i] = (d > 0.0f) ? rsqrtf(fmaxf(d, 1e-12f)) : 0.0f;
  }
}

// out[dst] += scale * dis[src]*dis[dst] * x[src]  (128-wide, one wave per edge)
__global__ __launch_bounds__(256) void prop128_kernel(
    const int* __restrict__ src, const int* __restrict__ dst,
    const float* __restrict__ dis, const float* __restrict__ x,
    float* __restrict__ out, float scale, int nE)
{
  int e = blockIdx.x * 8 + (threadIdx.x >> 5);
  if (e >= nE) return;
  int lane = threadIdx.x & 31;
  int s = src[e], d = dst[e];
  float c = scale * dis[s] * dis[d];
  const float4 xv = ((const float4*)(x + (size_t)s * 128))[lane];
  float* o = out + (size_t)d * 128 + lane * 4;
  atomicAdd(o + 0, c * xv.x);
  atomicAdd(o + 1, c * xv.y);
  atomicAdd(o + 2, c * xv.z);
  atomicAdd(o + 3, c * xv.w);
}

// 4-wide GCN edge prop: out[dst] += disg[s]*disg[d]*x[src]
__global__ void gprop4_kernel(const int* __restrict__ src, const int* __restrict__ dst,
                              const float* __restrict__ disg, const float* __restrict__ x,
                              float* __restrict__ out, int nE)
{
  int e = blockIdx.x * blockDim.x + threadIdx.x;
  if (e >= nE) return;
  int s = src[e], d = dst[e];
  float c = disg[s] * disg[d];
#pragma unroll
  for (int j = 0; j < 4; ++j)
    atomicAdd(&out[(size_t)d * 4 + j], c * x[(size_t)s * 4 + j]);
}

// W3eff[k][i][j] = sum_l sum_m cheb_w[l,k][i,m] * w3[l*128+m, j]
__global__ void w3eff_kernel(const float* __restrict__ chebw,
                             const float* __restrict__ w3, float* __restrict__ w3e)
{
  int gid = blockIdx.x * blockDim.x + threadIdx.x;
  if (gid >= 3 * 128 * 256) return;
  int j = gid & 255, i = (gid >> 8) & 127, k = gid >> 15;
  float s = 0.0f;
  for (int l = 0; l < 4; ++l) {
    const float* cw  = chebw + (((size_t)l * 3 + k) * 128 + i) * 128;
    const float* w3l = w3 + (size_t)l * 128 * 256 + j;
    for (int m = 0; m < 128; ++m) s += cw[m] * w3l[(size_t)m * 256];
  }
  w3e[gid] = s;
}
__global__ void beff_kernel(const float* __restrict__ chebb,
                            const float* __restrict__ w3,
                            const float* __restrict__ b3, float* __restrict__ beff)
{
  int j = blockIdx.x * blockDim.x + threadIdx.x;
  if (j >= 256) return;
  float s = b3[j];
  for (int l = 0; l < 4; ++l)
    for (int m = 0; m < 128; ++m)
      s += chebb[l * 128 + m] * w3[(size_t)(l * 128 + m) * 256 + j];
  beff[j] = s;
}

// scores[n] = <h4[n,:], t[batch[n],:]>  (256-wide dot, one wave per node)
__global__ __launch_bounds__(256) void scores_kernel(
    const float* __restrict__ h4, const float* __restrict__ t,
    const int* __restrict__ batch, float* __restrict__ scores, int n)
{
  int node = blockIdx.x * 8 + (threadIdx.x >> 5);
  if (node >= n) return;
  int lane = threadIdx.x & 31;
  int g = batch[node];
  const float* hr = h4 + (size_t)node * 256;
  const float* tr = t + (size_t)g * 256;
  float p = 0.0f;
  for (int j = lane; j < 256; j += 32) p += hr[j] * tr[j];
  for (int off = 16; off; off >>= 1) p += __shfl_xor(p, off, 32);
  if (lane == 0) scores[node] = p;
}

__global__ void counts_kernel(const int* __restrict__ batch, float* counts, int n) {
  int i = blockIdx.x * blockDim.x + threadIdx.x;
  if (i < n) atomicAdd(&counts[batch[i]], 1.0f);
}
__global__ void inv_kernel(const float* __restrict__ counts, float* invv, int g) {
  int i = blockIdx.x * blockDim.x + threadIdx.x;
  if (i < g) invv[i] = (counts[i] > 0.0f) ? 1.0f / counts[i] : 0.0f;
}

// hg[g] += scores[n]*inv[g] * h4[n]  (one wave per node)
__global__ __launch_bounds__(256) void hg_kernel(
    const float* __restrict__ h4, const float* __restrict__ scores,
    const float* __restrict__ invv, const int* __restrict__ batch,
    float* __restrict__ hg, int n)
{
  int node = blockIdx.x * 8 + (threadIdx.x >> 5);
  if (node >= n) return;
  int lane = threadIdx.x & 31;
  int g = batch[node];
  float w = scores[node] * invv[g];
  const float* hr = h4 + (size_t)node * 256;
  float* og = hg + (size_t)g * 256;
  for (int j = lane; j < 256; j += 32) atomicAdd(&og[j], w * hr[j]);
}

// xa = features @ g1w  (OUT=4: VALU)
__global__ void xa_kernel(const float* __restrict__ feat, const float* __restrict__ g1w,
                          float* __restrict__ xa, int n)
{
  int gid = blockIdx.x * blockDim.x + threadIdx.x;
  if (gid >= n * 4) return;
  int j = gid & 3, node = gid >> 2;
  const float* f = feat + (size_t)node * 128;
  float s = 0.0f;
  for (int m = 0; m < 128; ++m) s += f[m] * g1w[m * 4 + j];
  xa[gid] = s;
}
// x1 = relu(tmp4 + xa*disg^2 + g1b)
__global__ void x1_kernel(const float* __restrict__ tmp4, const float* __restrict__ xa,
                          const float* __restrict__ disg, const float* __restrict__ g1b,
                          float* __restrict__ x1, int n)
{
  int gid = blockIdx.x * blockDim.x + threadIdx.x;
  if (gid >= n * 4) return;
  int j = gid & 3, node = gid >> 2;
  float dg = disg[node];
  float v = tmp4[gid] + xa[gid] * dg * dg + g1b[j];
  x1[gid] = (v > 0.0f) ? v : 0.0f;
}
// xb = x1 @ g2w  ([N,4]x[4,128])
__global__ void xb_kernel(const float* __restrict__ x1, const float* __restrict__ g2w,
                          float* __restrict__ xb, int n)
{
  int gid = blockIdx.x * blockDim.x + threadIdx.x;
  if (gid >= n * 128) return;
  int j = gid & 127, node = gid >> 7;
  const float* xr = x1 + (size_t)node * 4;
  xb[gid] = xr[0] * g2w[j] + xr[1] * g2w[128 + j] + xr[2] * g2w[256 + j] + xr[3] * g2w[384 + j];
}
// emb[g] += (tmp128 + xb*disg^2 + g2b) * inv[g]
__global__ __launch_bounds__(256) void emb_kernel(
    const float* __restrict__ tmp128, const float* __restrict__ xb,
    const float* __restrict__ disg, const float* __restrict__ g2b,
    const float* __restrict__ invv, const int* __restrict__ batch,
    float* __restrict__ emb, int n)
{
  int node = blockIdx.x * 8 + (threadIdx.x >> 5);
  if (node >= n) return;
  int lane = threadIdx.x & 31;
  int g = batch[node];
  float dg = disg[node], d2 = dg * dg, iw = invv[g];
  const float* tr = tmp128 + (size_t)node * 128;
  const float* xr = xb + (size_t)node * 128;
  float* og = emb + (size_t)g * 128;
  for (int j = lane; j < 128; j += 32)
    atomicAdd(&og[j], (tr[j] + xr[j] * d2 + g2b[j]) * iw);
}

// out[g,c] = [hg|x2]@w7 + b7 + emb@pw + pb
__global__ void final_kernel(const float* __restrict__ hg, const float* __restrict__ x2,
                             const float* __restrict__ emb,
                             const float* __restrict__ w7, const float* __restrict__ b7,
                             const float* __restrict__ pw, const float* __restrict__ pb,
                             float* __restrict__ out)
{
  int gid = blockIdx.x * blockDim.x + threadIdx.x;
  if (gid >= GG * NCLSD) return;
  int c = gid & 1, g = gid >> 1;
  float s = b7[c] + pb[c];
  const float* hgr = hg + (size_t)g * 256;
  const float* x2r = x2 + (size_t)g * 256;
  for (int h = 0; h < 256; ++h) s += hgr[h] * w7[h * 2 + c];
  for (int h = 0; h < 256; ++h) s += x2r[h] * w7[(256 + h) * 2 + c];
  const float* er = emb + (size_t)g * 128;
  for (int d = 0; d < 128; ++d) s += er[d] * pw[d * 2 + c];
  out[gid] = s;
}

// ---------------------------------------------------------------------------
extern "C" void kernel_launch(void* const* d_in, const int* in_sizes, int n_in,
                              void* d_out, int out_size, void* d_ws, size_t ws_size,
                              hipStream_t stream) {
  (void)in_sizes; (void)n_in; (void)out_size; (void)ws_size;

  const float* features = (const float*)d_in[0];
  const int*   eidx     = (const int*)d_in[1];
  const int*   src      = eidx;
  const int*   dst      = eidx + EE;
  const int*   batch    = (const int*)d_in[2];
  const float* xLx      = (const float*)d_in[3];
  const float* w1 = (const float*)d_in[4];  const float* b1 = (const float*)d_in[5];
  const float* w2 = (const float*)d_in[6];  const float* b2 = (const float*)d_in[7];
  const float* chebw = (const float*)d_in[8];
  const float* chebb = (const float*)d_in[9];
  const float* w3 = (const float*)d_in[10]; const float* b3 = (const float*)d_in[11];
  const float* w4 = (const float*)d_in[12]; const float* b4 = (const float*)d_in[13];
  const float* w5 = (const float*)d_in[14]; const float* b5 = (const float*)d_in[15];
  const float* w6 = (const float*)d_in[16]; const float* b6 = (const float*)d_in[17];
  const float* w7 = (const float*)d_in[18]; const float* b7 = (const float*)d_in[19];
  const float* w8 = (const float*)d_in[20]; const float* b8 = (const float*)d_in[21];
  const float* w9 = (const float*)d_in[22]; const float* b9 = (const float*)d_in[23];
  const float* g1w = (const float*)d_in[24]; const float* g1b = (const float*)d_in[25];
  const float* g2w = (const float*)d_in[26]; const float* g2b = (const float*)d_in[27];
  const float* pw  = (const float*)d_in[28]; const float* pb  = (const float*)d_in[29];

  float* out = (float*)d_out;
  float* ws  = (float*)d_ws;

  const size_t N = NN;
  float* bufA   = ws;              // 128N : h1, later xb
  float* bufB   = ws + 128 * N;    // h ;  h4 occupies [128N, 384N)
  float* bufC   = ws + 256 * N;    // tx1, later tmp128
  float* bufD   = ws + 384 * N;    // tx2
  float* bufE   = ws + 512 * N;    // h3 (256N); later xa/x1/tmp4
  float* xa     = bufE;            // 4N
  float* x1b    = bufE + 4 * N;    // 4N
  float* tmp4   = bufE + 8 * N;    // 4N
  float* tail   = ws + 768 * N;
  float* deg    = tail;
  float* dis    = tail + N;
  float* degg   = tail + 2 * N;
  float* disg   = tail + 3 * N;
  float* scores = tail + 4 * N;
  float* small  = tail + 5 * N;
  float* w3e    = small;                 // 3*128*256
  float* beff   = w3e + 3 * 128 * 256;   // 256
  float* t1buf  = beff + 256;            // G*256
  float* tbuf   = t1buf + GG * 256;
  float* x2a    = tbuf + GG * 256;
  float* x2b    = x2a + GG * 256;
  float* hgb    = x2b + GG * 256;        // G*256
  float* embb   = hgb + GG * 256;        // G*128
  float* counts = embb + GG * 128;       // G
  float* invb   = counts + GG;           // G

  const float* NUL = nullptr;
  dim3 blk(256);
  auto gup = [](size_t n) { return (unsigned)((n + 255) / 256); };

  // --- degrees / normalizations ---
  zero_f32<<<gup(N), blk, 0, stream>>>(deg, N);
  fill_f32<<<gup(N), blk, 0, stream>>>(degg, 1.0f, N);  // self-loop
  scatter_deg<<<gup(EE), blk, 0, stream>>>(src, deg, EE);
  scatter_deg<<<gup(EE), blk, 0, stream>>>(dst, degg, EE);
  rsqrt_deg<<<gup(N), blk, 0, stream>>>(deg, dis, NN);
  rsqrt_deg<<<gup(N), blk, 0, stream>>>(degg, disg, NN);

  // --- intra: h1 = leaky(feat@w1+b1); h = leaky(h1@w2+b2)+h1 ---
  // node GEMMs: MT=5 (80 rows/block), 100000/80 = 1250 blocks
  gemm_wmma<128, 5, 1><<<NN / 80, blk, 0, stream>>>(features, w1, NUL, NUL, NUL, NUL,
      128, b1, NUL, bufA, 1);
  gemm_wmma<128, 5, 1><<<NN / 80, blk, 0, stream>>>(bufA, w2, NUL, NUL, NUL, NUL,
      128, b2, bufA, bufB, 1);

  // --- Cheb props: tx1 = -D^-1/2 A D^-1/2 h ; tx2 = 2*prop(tx1) - h ---
  zero_f32<<<gup(128 * N), blk, 0, stream>>>(bufC, 128 * N);
  prop128_kernel<<<EE / 8, blk, 0, stream>>>(src, dst, dis, bufB, bufC, -1.0f, EE);
  neg_copy<<<gup(128 * N), blk, 0, stream>>>(bufD, bufB, 128 * N);
  prop128_kernel<<<EE / 8, blk, 0, stream>>>(src, dst, dis, bufC, bufD, -2.0f, EE);

  // --- fold cheb_w and w3 into 3 effective [128,256] weights ---
  w3eff_kernel<<<gup(3 * 128 * 256), blk, 0, stream>>>(chebw, w3, w3e);
  beff_kernel<<<1, blk, 0, stream>>>(chebb, w3, b3, beff);

  // --- h3 = leaky(Tx0@W3e0 + Tx1@W3e1 + Tx2@W3e2 + beff) ; h4 = leaky(h3@w4+b4) ---
  gemm_wmma<256, 5, 3><<<NN / 80, blk, 0, stream>>>(bufB, w3e, bufC, w3e + 32768,
      bufD, w3e + 65536, 128, beff, NUL, bufE, 1);
  gemm_wmma<256, 5, 1><<<NN / 80, blk, 0, stream>>>(bufE, w4, NUL, NUL, NUL, NUL,
      256, b4, NUL, bufB, 1);  // h4 at bufB (256N)

  // --- graph-level small GEMMs (G=64 rows = 4 row tiles, 1 block) ---
  gemm_wmma<256, 4, 1><<<1, blk, 0, stream>>>(xLx, w8, NUL, NUL, NUL, NUL,
      128, b8, NUL, t1buf, 1);
  gemm_wmma<256, 4, 1><<<1, blk, 0, stream>>>(t1buf, w9, NUL, NUL, NUL, NUL,
      256, b9, NUL, tbuf, 1);
  gemm_wmma<256, 4, 1><<<1, blk, 0, stream>>>(xLx, w5, NUL, NUL, NUL, NUL,
      128, b5, NUL, x2a, 0);
  gemm_wmma<256, 4, 1><<<1, blk, 0, stream>>>(x2a, w6, NUL, NUL, NUL, NUL,
      256, b6, NUL, x2b, 1);

  // --- attention scores + pooling ---
  scores_kernel<<<gup(8 * N), blk, 0, stream>>>(bufB, tbuf, batch, scores, NN);
  zero_f32<<<1, blk, 0, stream>>>(counts, GG);
  counts_kernel<<<gup(N), blk, 0, stream>>>(batch, counts, NN);
  inv_kernel<<<1, blk, 0, stream>>>(counts, invb, GG);
  zero_f32<<<gup(GG * 256), blk, 0, stream>>>(hgb, GG * 256);
  hg_kernel<<<gup(8 * N), blk, 0, stream>>>(bufB, scores, invb, batch, hgb, NN);

  // --- inter analyzer (GCN x2 + mean pool) ---
  xa_kernel<<<gup(4 * N), blk, 0, stream>>>(features, g1w, xa, NN);
  zero_f32<<<gup(4 * N), blk, 0, stream>>>(tmp4, 4 * N);
  gprop4_kernel<<<gup(EE), blk, 0, stream>>>(src, dst, disg, xa, tmp4, EE);
  x1_kernel<<<gup(4 * N), blk, 0, stream>>>(tmp4, xa, disg, g1b, x1b, NN);
  xb_kernel<<<gup(128 * N), blk, 0, stream>>>(x1b, g2w, bufA, NN);
  zero_f32<<<gup(128 * N), blk, 0, stream>>>(bufC, 128 * N);  // tmp128
  prop128_kernel<<<EE / 8, blk, 0, stream>>>(src, dst, disg, bufA, bufC, 1.0f, EE);
  zero_f32<<<gup(GG * 128), blk, 0, stream>>>(embb, GG * 128);
  emb_kernel<<<gup(8 * N), blk, 0, stream>>>(bufC, bufA, disg, g2b, invb, batch, embb, NN);

  // --- final: intra + inter ---
  final_kernel<<<1, blk, 0, stream>>>(hgb, x2b, embb, w7, b7, pw, pb, out);
}
